// RegularisedTransport_17403207483481
// MI455X (gfx1250) — matrix operand order repaired and checked by reading, not compiled, not run on hardware
//
#include <hip/hip_runtime.h>

typedef __attribute__((ext_vector_type(16))) __bf16 v16bf;
typedef __attribute__((ext_vector_type(8)))  float  v8f;

#define B_      8
#define MN      1024
#define ITERS   100
#define TPB     256
#define COLS    64                      // columns (and rows) owned by each WG
#define WGS_PER_B 16
#define TILE_BYTES (4*32*1024)          // 4 groups x 32 k-chunks x 1KB tiles
#define ABUF_BYTES 2048                 // 32 chunks x 2 lane-images x 32B (bf16 A-layout)
#define PART_BYTES 2048                 // 8 waves x 4 groups x 16 partials (f32)
#define LDS_BYTES  (2*TILE_BYTES + ABUF_BYTES + PART_BYTES)   // 266,240 B < 320KB

#define WMMA_BF16(A, B, C) \
    __builtin_amdgcn_wmma_f32_16x16x32_bf16(false, (A), false, (B), \
                                            (short)0, (C), false, false)

__device__ __forceinline__ float kval(float m) {
    return __expf(-10.0f * fminf(m, 5.0f));
}

// Pack a 1024-long f32 vector (optionally x -> numer/x) into WMMA A-operand
// bf16 layout: per 32-chunk, two 32B lane images (lanes<16: K 0-15;
// lanes>=16: K 16-31), rows replicated so image is per-half-wave.
__device__ __forceinline__ void build_abuf(unsigned* abuf, const float* src,
                                           float numer, bool recip, int tid) {
    for (int i = tid; i < 512; i += TPB) {
        int c = i >> 4, h = (i >> 3) & 1, w = i & 7;
        int k0 = h * 16 + 2 * w;
        float f0 = src[(c << 5) + k0];
        float f1 = src[(c << 5) + k0 + 1];
        if (recip) { f0 = numer / f0; f1 = numer / f1; }
        unsigned short lo = __builtin_bit_cast(unsigned short, (__bf16)f0);
        unsigned short hi = __builtin_bit_cast(unsigned short, (__bf16)f1);
        abuf[i] = ((unsigned)hi << 16) | lo;
    }
}

// One mat-vec phase: wave w covers k-chunks 4w..4w+3 over all 4 column groups.
// Software-pipelined: chunk q+1's A + 4 B tiles are in flight across chunk q's
// 4 WMMAs, so ds_load latency overlaps matrix issue.
__device__ __forceinline__ void mv_phase(const char* __restrict__ tiles,
                                         const unsigned* __restrict__ abuf,
                                         float* __restrict__ part,
                                         int w, int lane, int aoff) {
    const int c0 = w << 2;
    v8f acc0 = {}, acc1 = {}, acc2 = {}, acc3 = {};

    v16bf A  = *(const v16bf*)((const char*)abuf + (c0 << 6) + aoff);
    const char* bb = tiles + (c0 << 10) + (lane << 5);
    v16bf B0 = *(const v16bf*)(bb);
    v16bf B1 = *(const v16bf*)(bb + (1 << 15));
    v16bf B2 = *(const v16bf*)(bb + (2 << 15));
    v16bf B3 = *(const v16bf*)(bb + (3 << 15));

    #pragma unroll
    for (int q = 0; q < 4; ++q) {
        v16bf An, N0, N1, N2, N3;
        if (q < 3) {                               // prefetch chunk q+1
            const int cn = c0 + q + 1;
            An = *(const v16bf*)((const char*)abuf + (cn << 6) + aoff);
            const char* nb = tiles + (cn << 10) + (lane << 5);
            N0 = *(const v16bf*)(nb);
            N1 = *(const v16bf*)(nb + (1 << 15));
            N2 = *(const v16bf*)(nb + (2 << 15));
            N3 = *(const v16bf*)(nb + (3 << 15));
        }
        acc0 = WMMA_BF16(A, B0, acc0);
        acc1 = WMMA_BF16(A, B1, acc1);
        acc2 = WMMA_BF16(A, B2, acc2);
        acc3 = WMMA_BF16(A, B3, acc3);
        if (q < 3) { A = An; B0 = N0; B1 = N1; B2 = N2; B3 = N3; }
    }
    if (lane < 16) {                               // D row 0 (all rows identical)
        part[(w << 6) +  0 + lane] = acc0[0];
        part[(w << 6) + 16 + lane] = acc1[0];
        part[(w << 6) + 32 + lane] = acc2[0];
        part[(w << 6) + 48 + lane] = acc3[0];
    }
}

// Persistent per-batch-slice Sinkhorn iteration kernel.
// grid = 8 batches * 16 WGs; WG j of batch b owns columns/rows [64j, 64j+64).
__global__ void __launch_bounds__(TPB)
sinkhorn_iter(const float* __restrict__ Mg, float* __restrict__ ws) {
    extern __shared__ v16bf smem_v[];          // 32B-aligned dynamic LDS
    char* lds = (char*)smem_v;

    const int tid = threadIdx.x;
    const int b   = blockIdx.x >> 4;
    const int wg  = blockIdx.x & 15;
    const int n0  = wg * COLS;
    const float* Mb = Mg + (size_t)b * MN * MN;

    float* y_g = ws;                           // [B][1024]
    float* u_g = ws + B_ * MN;                 // [B][1024]
    float* v_g = ws + 2 * B_ * MN;             // [B][1024]
    unsigned* cnt = (unsigned*)(ws + 3 * B_ * MN) + b * 64;

    char*     t1   = lds;                      // phase-1 B-tiles (K column slice)
    char*     t2   = lds + TILE_BYTES;         // phase-2 B-tiles (K row slice, transposed)
    unsigned* abuf = (unsigned*)(lds + 2 * TILE_BYTES);
    float*    part = (float*)(lds + 2 * TILE_BYTES + ABUF_BYTES);

    const float Rm = 1.0f / MN;
    const float Cn = 1.0f / MN;

    // ---- one-time staging: recompute K from M, pack bf16 WMMA B-tiles ----
    // Phase-1 tile (g,ch): B[k][j] = K[m = ch*32+k][n = n0 + g*16 + j]
    for (int i = tid; i < MN * COLS; i += TPB) {
        int ncol = i & (COLS - 1);
        int m    = i >> 6;
        float kv = kval(Mb[(size_t)m * MN + n0 + ncol]);
        int g = ncol >> 4, jj = ncol & 15, ch = m >> 5, k = m & 31;
        int l = (k & 16) + jj, e = k & 15;
        *(__bf16*)(t1 + (((g << 5) + ch) << 10) + (l << 5) + (e << 1)) = (__bf16)kv;
    }
    // Phase-2 tile (g,ch): B[k][j] = K[m = n0 + g*16 + j][n = ch*32+k]
    for (int i = tid; i < COLS * MN; i += TPB) {
        int nn   = i & (MN - 1);
        int mrow = i >> 10;
        float kv = kval(Mb[(size_t)(n0 + mrow) * MN + nn]);
        int g = mrow >> 4, jj = mrow & 15, ch = nn >> 5, k = nn & 31;
        int l = (k & 16) + jj, e = k & 15;
        *(__bf16*)(t2 + (((g << 5) + ch) << 10) + (l << 5) + (e << 1)) = (__bf16)kv;
    }
    {   // u0 = r everywhere (constant bf16 image)
        unsigned short hs = __builtin_bit_cast(unsigned short, (__bf16)Rm);
        unsigned pk = ((unsigned)hs << 16) | hs;
        for (int i = tid; i < 512; i += TPB) abuf[i] = pk;
    }
    __syncthreads();

    const int w    = tid >> 5, lane = tid & 31;
    const int aoff = (lane & 16) << 1;         // 0 or 32: lane-image select

    unsigned sidx = 0;

    for (int iter = 0; iter <= ITERS; ++iter) {
        // ---------- phase 1: y = K^T u  (LDS-resident, WMMA) ----------
        mv_phase(t1, abuf, part, w, lane, aoff);
        __syncthreads();

        float red = 0.0f;
        if (tid < COLS) {
            #pragma unroll
            for (int w2 = 0; w2 < 8; ++w2) red += part[(w2 << 6) + tid];
        }

        if (iter == ITERS) {                   // v = c / (K^T u_final)
            if (tid < COLS) v_g[(b << 10) + n0 + tid] = Cn / red;
            return;
        }

        if (tid < COLS) y_g[(b << 10) + n0 + tid] = red;
        __threadfence();
        __syncthreads();
        ++sidx;
        if (tid == 0) {
            atomicAdd(cnt, 1u);
            while (__hip_atomic_load(cnt, __ATOMIC_ACQUIRE, __HIP_MEMORY_SCOPE_AGENT) < 16u * sidx)
                __builtin_amdgcn_s_sleep(2);
        }
        __syncthreads();
        __threadfence();
        build_abuf(abuf, y_g + (b << 10), Cn, true, tid);   // A = c / y
        __syncthreads();

        // ---------- phase 2: s = K (c/y) ; u = r / s ----------
        mv_phase(t2, abuf, part, w, lane, aoff);
        __syncthreads();
        if (tid < COLS) {
            float s = 0.0f;
            #pragma unroll
            for (int w2 = 0; w2 < 8; ++w2) s += part[(w2 << 6) + tid];
            u_g[(b << 10) + n0 + tid] = Rm / s;
        }
        __threadfence();
        __syncthreads();
        ++sidx;
        if (tid == 0) {
            atomicAdd(cnt, 1u);
            while (__hip_atomic_load(cnt, __ATOMIC_ACQUIRE, __HIP_MEMORY_SCOPE_AGENT) < 16u * sidx)
                __builtin_amdgcn_s_sleep(2);
        }
        __syncthreads();
        __threadfence();
        build_abuf(abuf, u_g + (b << 10), 0.0f, false, tid);  // A = u
        __syncthreads();
    }
}

// P = u * K * v^T, K recomputed from M at full f32 precision; float4 streams.
__global__ void __launch_bounds__(TPB)
sinkhorn_P(const float* __restrict__ Mg, const float* __restrict__ ws,
           float* __restrict__ P) {
    const float* u_g = ws + B_ * MN;
    const float* v_g = ws + 2 * B_ * MN;
    size_t idx = (size_t)blockIdx.x * TPB + threadIdx.x;   // one float4 each
    int    n4  = (int)(idx & 255);
    size_t row = idx >> 8;                                 // = b*1024 + m
    int    b   = (int)(row >> 10);
    float  u   = u_g[row];
    float4 v4  = ((const float4*)(v_g + ((size_t)b << 10)))[n4];
    float4 m4  = ((const float4*)Mg)[idx];
    float4 o;
    o.x = u * kval(m4.x) * v4.x;
    o.y = u * kval(m4.y) * v4.y;
    o.z = u * kval(m4.z) * v4.z;
    o.w = u * kval(m4.w) * v4.w;
    ((float4*)P)[idx] = o;
}

extern "C" void kernel_launch(void* const* d_in, const int* in_sizes, int n_in,
                              void* d_out, int out_size, void* d_ws, size_t ws_size,
                              hipStream_t stream) {
    const float* Mg = (const float*)d_in[0];
    float* ws = (float*)d_ws;

    // Reset per-batch sync counters every call (graph-capture-safe memset node).
    hipMemsetAsync((char*)d_ws + (size_t)3 * B_ * MN * sizeof(float), 0,
                   B_ * 64 * sizeof(unsigned), stream);

    hipFuncSetAttribute((const void*)sinkhorn_iter,
                        hipFuncAttributeMaxDynamicSharedMemorySize, LDS_BYTES);

    hipLaunchKernelGGL(sinkhorn_iter, dim3(B_ * WGS_PER_B), dim3(TPB),
                       LDS_BYTES, stream, Mg, ws);

    const int p_blocks = (B_ * MN * MN / 4) / TPB;   // 8192
    hipLaunchKernelGGL(sinkhorn_P, dim3(p_blocks), dim3(TPB), 0, stream,
                       Mg, ws, (float*)d_out);
}
